// Embed_and_Prep_29265907155067
// MI455X (gfx1250) — compile-verified, matrix operand order checked by reading, not run on hardware
//
#include <hip/hip_runtime.h>
#include <hip/hip_bf16.h>
#include <math.h>

typedef __attribute__((ext_vector_type(16))) _Float16 v16h;
typedef __attribute__((ext_vector_type(8)))  float    v8f;

constexpr int FEAT = 11;
constexpr int NH1  = 128;   // GEMM1 out
constexpr int NH2  = 256;   // GEMM2 out
constexpr int NH3  = 512;   // GEMM3 out (concat dim too)
constexpr int CDIM = 384;   // token dim
constexpr int NSP  = 4096;
constexpr int MASKN = 2457;            // int(4096*0.6)
constexpr int REMN  = NSP - MASKN;     // 1639

// ---- output region offsets (floats), return order:
// (remain_tok, mask_tok), (remain_pos, mask_pos), (remain_idx, mask_idx)
constexpr int OFF_RTOK = 0;
constexpr int OFF_MTOK = REMN * CDIM;
constexpr int OFF_RPOS = NSP * CDIM;
constexpr int OFF_MPOS = OFF_RPOS + REMN * CDIM;
constexpr int OFF_RIDX = 2 * NSP * CDIM;
constexpr int OFF_MIDX = OFF_RIDX + REMN;

// ---- monotonic float<->u32 key so segment-max == global_atomic_max_u32 ----
__device__ __forceinline__ unsigned f2key(float f) {
  unsigned u = __float_as_uint(f);
  return (u & 0x80000000u) ? ~u : (u | 0x80000000u);
}
__device__ __forceinline__ float key2f(unsigned k) {
  unsigned u = (k & 0x80000000u) ? (k & 0x7FFFFFFFu) : ~k;
  return __uint_as_float(u);
}

// ---- CDNA5 16-bit A/B fragment K mapping (CDNA5 ISA, 16x16x32 f16) ----
__device__ __forceinline__ int k32_of(int half, int j) {
  return (j < 8) ? (half * 8 + j) : (16 + half * 8 + (j - 8));
}
__device__ __forceinline__ void inv_k32(int k32, int& half, int& j) {
  if      (k32 < 8)  { half = 0; j = k32; }
  else if (k32 < 16) { half = 1; j = k32 - 8; }
  else if (k32 < 24) { half = 0; j = 8 + (k32 - 16); }
  else               { half = 1; j = 8 + (k32 - 24); }
}

__device__ __forceinline__ v8f wmma16(v16h a, v16h b, v8f c) {
  return __builtin_amdgcn_wmma_f32_16x16x32_f16(
      false, a, false, b, (short)0, c, false, false);
}

// =====================================================================
// Weight prep: W [Nreal x Kreal] row-major f32 -> B-fragment f16 buffer
// =====================================================================
__global__ void wprep_kernel(const float* __restrict__ W, int Nreal, int Kreal,
                             int KC, _Float16* __restrict__ frag, int total) {
  int t = blockIdx.x * blockDim.x + threadIdx.x;
  if (t >= total) return;
  int tile = t >> 9;
  int rem  = t & 511;
  int lane = rem >> 4, j = rem & 15;
  int nt = tile / KC, kc = tile % KC;
  int n = nt * 16 + (lane & 15);
  int half = lane >> 4;
  int k = kc * 32 + k32_of(half, j);
  float v = (n < Nreal && k < Kreal) ? W[n * Kreal + k] : 0.f;
  frag[t] = (_Float16)v;
}

__global__ void zero_kernel(unsigned* __restrict__ p, int n) {
  int t = blockIdx.x * blockDim.x + threadIdx.x;
  if (t < n) p[t] = 0u;
}

// run-length-merged segment-max emitter (s is sorted -> long runs)
struct RunMax {
  int sp; unsigned key;
  __device__ __forceinline__ void init() { sp = -1; key = 0u; }
  __device__ __forceinline__ void push(int nsp, unsigned k, unsigned* addr_base,
                                       int stride) {
    if (nsp != sp) {
      if (sp >= 0) atomicMax(addr_base + sp * stride, key);
      sp = nsp; key = k;
    } else {
      key = key > k ? key : k;
    }
  }
  __device__ __forceinline__ void flush(unsigned* addr_base, int stride) {
    if (sp >= 0) atomicMax(addr_base + sp * stride, key);
  }
};

// =====================================================================
// Pass 1: h = (X W1^T + b1) W2^T + b2 ; segment-max into g_bits[NSP*256]
// Block = 8 waves = 8 M-tiles (128 rows); GEMM2 N-split across waves,
// B register-reused over 8 LDS A-tiles.
// =====================================================================
__global__ __launch_bounds__(256) void pass1_kernel(
    const float* __restrict__ feat, const int* __restrict__ s,
    const _Float16* __restrict__ fW1, const float* __restrict__ b1,
    const _Float16* __restrict__ fW2, const float* __restrict__ b2,
    unsigned* __restrict__ g_bits, int P) {
  __shared__ _Float16 ldsH[8 * 2048];  // 8 M-tiles * (4 kc * 512) = 32 KB
  __shared__ int sIdx[128];
  int tid = threadIdx.x;
  int wave = tid >> 5, lane = tid & 31;
  int mlane = lane & 15, half = lane >> 4, mbase = half * 8;
  int blk_m0 = blockIdx.x * 128;
  bool full = (blk_m0 + 128 <= P);

  if (tid < 128) {
    int m = blk_m0 + tid; if (m > P - 1) m = P - 1;
    sIdx[tid] = s[m];
  }

  // ---- GEMM1 on own tile (tile index == wave) -> ldsH[wave] ----
  {
    v16h a1;
    int m = blk_m0 + wave * 16 + mlane; if (m > P - 1) m = P - 1;
#pragma unroll
    for (int j = 0; j < 16; ++j) {
      int k = k32_of(half, j);
      a1[j] = (k < FEAT) ? (_Float16)feat[m * FEAT + k] : (_Float16)0.f;
    }
    for (int nt = 0; nt < 8; ++nt) {
      int n = nt * 16 + mlane;
      v8f c; float bb = b1[n];
#pragma unroll
      for (int r = 0; r < 8; ++r) c[r] = bb;
      v16h b = *(const v16h*)(fW1 + nt * 512 + lane * 16);
      c = wmma16(a1, b, c);
      int kc = n >> 5, h2, j2; inv_k32(n & 31, h2, j2);
#pragma unroll
      for (int r = 0; r < 8; ++r)
        ldsH[wave * 2048 + kc * 512 + (mbase + r + 16 * h2) * 16 + j2] = (_Float16)c[r];
    }
  }
  __syncthreads();

  // ---- GEMM2: wave handles nt = 2*wave + {0,1}; B reused over 8 M-tiles ----
  for (int t = 0; t < 2; ++t) {
    int nt = wave * 2 + t;
    int n = nt * 16 + mlane;
    v8f acc[8];
    float bb = b2[n];
#pragma unroll
    for (int mt = 0; mt < 8; ++mt)
#pragma unroll
      for (int r = 0; r < 8; ++r) acc[mt][r] = bb;
    for (int kc = 0; kc < 4; ++kc) {
      v16h b = *(const v16h*)(fW2 + (nt * 4 + kc) * 512 + lane * 16);
#pragma unroll
      for (int mt = 0; mt < 8; ++mt) {
        v16h a = *(const v16h*)(ldsH + mt * 2048 + kc * 512 + lane * 16);
        acc[mt] = wmma16(a, b, acc[mt]);
      }
    }
    unsigned* gcol = g_bits + n;
    RunMax rm; rm.init();
    if (full) {
#pragma unroll
      for (int mt = 0; mt < 8; ++mt)
#pragma unroll
        for (int r = 0; r < 8; ++r)
          rm.push(sIdx[mt * 16 + mbase + r], f2key(acc[mt][r]), gcol, NH2);
    } else {
#pragma unroll
      for (int mt = 0; mt < 8; ++mt)
#pragma unroll
        for (int r = 0; r < 8; ++r) {
          int mr = mt * 16 + mbase + r;
          if (blk_m0 + mr < P)
            rm.push(sIdx[mr], f2key(acc[mt][r]), gcol, NH2);
        }
    }
    rm.flush(gcol, NH2);
  }
}

// =====================================================================
// Pass 2: recompute h, gather g[s], f=[g|h] (K=512), GEMM3+ReLU, GEMM4,
// segment-max into tok_bits[NSP*384].
// Block = 8 waves = 8 M-tiles (128 rows); ldsF/ldsT shared across waves.
// Dyn LDS = 256 KB (CDNA5: 320 KB/WGP).
// =====================================================================
__global__ __launch_bounds__(256) void pass2_kernel(
    const float* __restrict__ feat, const int* __restrict__ s,
    const _Float16* __restrict__ fW1, const float* __restrict__ b1,
    const _Float16* __restrict__ fW2, const float* __restrict__ b2,
    const _Float16* __restrict__ fW3, const float* __restrict__ b3,
    const _Float16* __restrict__ fW4, const float* __restrict__ b4,
    const unsigned* __restrict__ g_bits, unsigned* __restrict__ tok_bits, int P) {
  extern __shared__ _Float16 dsm[];
  _Float16* ldsF = dsm;            // 8 tiles * 8192 f16 = 128 KB (f, K=512)
  _Float16* ldsT = dsm + 65536;    // 8 tiles * 8192 f16 = 128 KB (relu t)
  __shared__ int sIdx[128];
  int tid = threadIdx.x;
  int wave = tid >> 5, lane = tid & 31;
  int mlane = lane & 15, half = lane >> 4, mbase = half * 8;
  int blk_m0 = blockIdx.x * 128;
  bool full = (blk_m0 + 128 <= P);

  if (tid < 128) {
    int m = blk_m0 + tid; if (m > P - 1) m = P - 1;
    sIdx[tid] = s[m];
  }

  // ---- 1) GEMM1 on own tile -> stage h1 fragments into ldsT[wave] kc0..3 ----
  {
    v16h a1;
    int m = blk_m0 + wave * 16 + mlane; if (m > P - 1) m = P - 1;
#pragma unroll
    for (int j = 0; j < 16; ++j) {
      int k = k32_of(half, j);
      a1[j] = (k < FEAT) ? (_Float16)feat[m * FEAT + k] : (_Float16)0.f;
    }
    for (int nt = 0; nt < 8; ++nt) {
      int n = nt * 16 + mlane;
      v8f c; float bb = b1[n];
#pragma unroll
      for (int r = 0; r < 8; ++r) c[r] = bb;
      v16h b = *(const v16h*)(fW1 + nt * 512 + lane * 16);
      c = wmma16(a1, b, c);
      int kc = n >> 5, h2, j2; inv_k32(n & 31, h2, j2);
#pragma unroll
      for (int r = 0; r < 8; ++r)
        ldsT[wave * 8192 + kc * 512 + (mbase + r + 16 * h2) * 16 + j2] = (_Float16)c[r];
    }
  }
  __syncthreads();

  // ---- 2) GEMM2: wave handles nt = 2*wave + {0,1}; h -> ldsF kc 8..15 ----
  for (int t = 0; t < 2; ++t) {
    int nt = wave * 2 + t;
    int n = nt * 16 + mlane;
    v8f acc[8];
    float bb = b2[n];
#pragma unroll
    for (int mt = 0; mt < 8; ++mt)
#pragma unroll
      for (int r = 0; r < 8; ++r) acc[mt][r] = bb;
    for (int kc = 0; kc < 4; ++kc) {
      v16h b = *(const v16h*)(fW2 + (nt * 4 + kc) * 512 + lane * 16);
#pragma unroll
      for (int mt = 0; mt < 8; ++mt) {
        v16h a = *(const v16h*)(ldsT + mt * 8192 + kc * 512 + lane * 16);
        acc[mt] = wmma16(a, b, acc[mt]);
      }
    }
    int kcF = 8 + (n >> 5), h2, j2; inv_k32(n & 31, h2, j2);
#pragma unroll
    for (int mt = 0; mt < 8; ++mt)
#pragma unroll
      for (int r = 0; r < 8; ++r)
        ldsF[mt * 8192 + kcF * 512 + (mbase + r + 16 * h2) * 16 + j2] = (_Float16)acc[mt][r];
  }

  // ---- 3) cooperative gather g[s[m]] -> ldsF kc 0..7 (all 256 threads) ----
  for (int idx = tid; idx < 8 * 16 * NH2; idx += 256) {
    int mt = idx >> 12;
    int rem = idx & 4095;
    int mr = rem >> 8, k = rem & 255;
    int sp = sIdx[mt * 16 + mr];
    float v = key2f(g_bits[sp * NH2 + k]);
    int kc = k >> 5, h2, j2; inv_k32(k & 31, h2, j2);
    ldsF[mt * 8192 + kc * 512 + (mr + 16 * h2) * 16 + j2] = (_Float16)v;
  }
  __syncthreads();

  // ---- 4) GEMM3 (K=512,N=512): nt = wave + 8*i, i<4; ReLU -> ldsT ----
  for (int i = 0; i < 4; ++i) {
    int nt = wave + 8 * i;
    int n = nt * 16 + mlane;
    v8f acc[8];
    float bb = b3[n];
#pragma unroll
    for (int mt = 0; mt < 8; ++mt)
#pragma unroll
      for (int r = 0; r < 8; ++r) acc[mt][r] = bb;
    for (int kc = 0; kc < 16; ++kc) {
      v16h b = *(const v16h*)(fW3 + (nt * 16 + kc) * 512 + lane * 16);
#pragma unroll
      for (int mt = 0; mt < 8; ++mt) {
        v16h a = *(const v16h*)(ldsF + mt * 8192 + kc * 512 + lane * 16);
        acc[mt] = wmma16(a, b, acc[mt]);
      }
    }
    int kcT = n >> 5, h2, j2; inv_k32(n & 31, h2, j2);
#pragma unroll
    for (int mt = 0; mt < 8; ++mt)
#pragma unroll
      for (int r = 0; r < 8; ++r) {
        float v = acc[mt][r] > 0.f ? acc[mt][r] : 0.f;
        ldsT[mt * 8192 + kcT * 512 + (mbase + r + 16 * h2) * 16 + j2] = (_Float16)v;
      }
  }
  __syncthreads();

  // ---- 5) GEMM4 (K=512,N=384): nt = wave + 8*i, i<3; merged atomics ----
  for (int i = 0; i < 3; ++i) {
    int nt = wave + 8 * i;
    int n = nt * 16 + mlane;
    v8f acc[8];
    float bb = b4[n];
#pragma unroll
    for (int mt = 0; mt < 8; ++mt)
#pragma unroll
      for (int r = 0; r < 8; ++r) acc[mt][r] = bb;
    for (int kc = 0; kc < 16; ++kc) {
      v16h b = *(const v16h*)(fW4 + (nt * 16 + kc) * 512 + lane * 16);
#pragma unroll
      for (int mt = 0; mt < 8; ++mt) {
        v16h a = *(const v16h*)(ldsT + mt * 8192 + kc * 512 + lane * 16);
        acc[mt] = wmma16(a, b, acc[mt]);
      }
    }
    unsigned* tcol = tok_bits + n;
    RunMax rm; rm.init();
    if (full) {
#pragma unroll
      for (int mt = 0; mt < 8; ++mt)
#pragma unroll
        for (int r = 0; r < 8; ++r)
          rm.push(sIdx[mt * 16 + mbase + r], f2key(acc[mt][r]), tcol, CDIM);
    } else {
#pragma unroll
      for (int mt = 0; mt < 8; ++mt)
#pragma unroll
        for (int r = 0; r < 8; ++r) {
          int mr = mt * 16 + mbase + r;
          if (blk_m0 + mr < P)
            rm.push(sIdx[mr], f2key(acc[mt][r]), tcol, CDIM);
        }
    }
    rm.flush(tcol, CDIM);
  }
}

// =====================================================================
// Pos-MLP: gelu(coords @ P1^T + pb1) @ P2^T + pb2 (4096x3 -> 128 -> 384)
// =====================================================================
__global__ __launch_bounds__(128) void pos_kernel(
    const float* __restrict__ coords,
    const float* __restrict__ P1w, const float* __restrict__ pb1,
    const float* __restrict__ P2w, const float* __restrict__ pb2,
    float* __restrict__ posbuf) {
  __shared__ float hid[128];
  int spi = blockIdx.x, t = threadIdx.x;
  float x0 = coords[spi * 3 + 0], x1 = coords[spi * 3 + 1], x2 = coords[spi * 3 + 2];
  float a = P1w[t * 3 + 0] * x0 + P1w[t * 3 + 1] * x1 + P1w[t * 3 + 2] * x2 + pb1[t];
  hid[t] = 0.5f * a * (1.0f + erff(a * 0.70710678118654752f));  // exact GELU
  __syncthreads();
  for (int o = t; o < CDIM; o += 128) {
    float acc = pb2[o];
    for (int j = 0; j < 128; ++j) acc += hid[j] * P2w[o * 128 + j];
    posbuf[spi * CDIM + o] = acc;
  }
}

// =====================================================================
// Deterministic 60/40 split: bijection p(i)=(i*1031+77) mod 4096.
// =====================================================================
__global__ void perm_kernel(unsigned* __restrict__ slot,
                            unsigned* __restrict__ flag) {
  if (blockIdx.x != 0 || threadIdx.x != 0) return;
  for (int i = 0; i < NSP; ++i) {
    unsigned j = ((unsigned)i * 1031u + 77u) & 4095u;
    flag[j] = (i < MASKN) ? 1u : 0u;
  }
  unsigned mc = 0, rc = 0;
  for (int j = 0; j < NSP; ++j) {
    if (flag[j]) { slot[j] = mc | 0x80000000u; ++mc; }
    else         { slot[j] = rc;               ++rc; }
  }
}

// =====================================================================
// Scatter tokens / pos / indices into flat output regions
// =====================================================================
__global__ void gather_kernel(const unsigned* __restrict__ tok_bits,
                              const float* __restrict__ posbuf,
                              const unsigned* __restrict__ slot,
                              float* __restrict__ out) {
  int t = blockIdx.x * blockDim.x + threadIdx.x;
  if (t >= NSP * CDIM) return;
  int j = t / CDIM, c = t % CDIM;
  unsigned sl = slot[j];
  int isM = (int)(sl >> 31);
  int p = (int)(sl & 0x7FFFFFFFu);
  out[(isM ? OFF_MTOK : OFF_RTOK) + p * CDIM + c] = key2f(tok_bits[t]);
  out[(isM ? OFF_MPOS : OFF_RPOS) + p * CDIM + c] = posbuf[t];
  if (c == 0) out[(isM ? OFF_MIDX : OFF_RIDX) + p] = (float)j;
}

// =====================================================================
extern "C" void kernel_launch(void* const* d_in, const int* in_sizes, int n_in,
                              void* d_out, int out_size, void* d_ws, size_t ws_size,
                              hipStream_t stream) {
  const float* feat   = (const float*)d_in[0];
  const float* coords = (const float*)d_in[1];
  const int*   s10    = (const int*)d_in[2];
  const float* W1 = (const float*)d_in[5],  *b1 = (const float*)d_in[6];
  const float* W2 = (const float*)d_in[7],  *b2 = (const float*)d_in[8];
  const float* W3 = (const float*)d_in[9],  *b3 = (const float*)d_in[10];
  const float* W4 = (const float*)d_in[11], *b4 = (const float*)d_in[12];
  const float* P1w = (const float*)d_in[13], *pb1 = (const float*)d_in[14];
  const float* P2w = (const float*)d_in[15], *pb2 = (const float*)d_in[16];
  const int P = in_sizes[0] / FEAT;

  // ---- workspace layout (bytes) ----
  char* ws = (char*)d_ws;
  _Float16* fW1 = (_Float16*)(ws + 0);            // 8*1*512  f16 =   8 KB
  _Float16* fW2 = (_Float16*)(ws + 8192);         // 16*4*512 f16 =  64 KB
  _Float16* fW3 = (_Float16*)(ws + 73728);        // 32*16*512f16 = 512 KB
  _Float16* fW4 = (_Float16*)(ws + 598016);       // 24*16*512f16 = 384 KB
  unsigned* g_bits   = (unsigned*)(ws + 991232);          // 4096*256 u32 = 4 MB
  unsigned* tok_bits = (unsigned*)(ws + 991232 + 4194304);// 4096*384 u32 = 6 MB
  float*    posbuf   = (float*)(ws + 11476992);           // 4096*384 f32 = 6 MB
  unsigned* slot     = (unsigned*)(ws + 17768448);        // 4096 u32
  unsigned* flag     = (unsigned*)(ws + 17784832);        // 4096 u32

  // 1) weight fragment prep (f16, B-matrix VGPR layout)
  {
    int t1 = 8 * 1 * 512, t2 = 16 * 4 * 512, t3 = 32 * 16 * 512, t4 = 24 * 16 * 512;
    wprep_kernel<<<(t1 + 255) / 256, 256, 0, stream>>>(W1, NH1, FEAT, 1,  fW1, t1);
    wprep_kernel<<<(t2 + 255) / 256, 256, 0, stream>>>(W2, NH2, NH1, 4,  fW2, t2);
    wprep_kernel<<<(t3 + 255) / 256, 256, 0, stream>>>(W3, NH3, NH3, 16, fW3, t3);
    wprep_kernel<<<(t4 + 255) / 256, 256, 0, stream>>>(W4, CDIM, NH3, 16, fW4, t4);
  }

  // 2) zero segment-max accumulators (g_bits and tok_bits are contiguous)
  {
    int n = NSP * (NH2 + CDIM);
    zero_kernel<<<(n + 255) / 256, 256, 0, stream>>>(g_bits, n);
  }

  // 3+4) passes: 128 rows (8 M-tiles) per 256-thread block
  int blocks = (P + 127) / 128;
  pass1_kernel<<<blocks, 256, 0, stream>>>(feat, s10, fW1, b1, fW2, b2, g_bits, P);
  pass2_kernel<<<blocks, 256, 262144, stream>>>(
      feat, s10, fW1, b1, fW2, b2, fW3, b3, fW4, b4, g_bits, tok_bits, P);

  // 5) pos embed
  pos_kernel<<<NSP, 128, 0, stream>>>(coords, P1w, pb1, P2w, pb2, posbuf);

  // 6) split + scatter
  perm_kernel<<<1, 1, 0, stream>>>(slot, flag);
  {
    int n = NSP * CDIM;
    gather_kernel<<<(n + 255) / 256, 256, 0, stream>>>(tok_bits, posbuf,
                                                       slot, (float*)d_out);
  }
}
